// WHISPER_DECODER_5205500362790
// MI455X (gfx1250) — matrix operand order repaired and checked by reading, not compiled, not run on hardware
//
#include <hip/hip_runtime.h>
#include <hip/hip_bf16.h>

// ---------------- CDNA5 WMMA types ----------------
typedef __attribute__((ext_vector_type(16))) __bf16 bf16x16;
typedef __attribute__((ext_vector_type(8)))  __bf16 bf16x8;
typedef __attribute__((ext_vector_type(8)))  float  floatx8;
typedef int v4i_t __attribute__((vector_size(16)));   // b128 payload type

#define S_LEN 448
#define D_DIM 768
#define H_NUM 12
#define HD_DIM 64
#define FF_DIM 3072
#define V_SIZE 51865
#define TENC_LEN 1500
#define L_NUM 12

// gfx1250 async global->LDS path (ASYNCcnt), compile-time guarded
#if __has_builtin(__builtin_amdgcn_global_load_async_to_lds_b128) && \
    __has_builtin(__builtin_amdgcn_s_wait_asynccnt)
#define HAVE_ASYNC_LDS 1
#else
#define HAVE_ASYNC_LDS 0
#endif

// =====================================================================
// Batched GEMM: C[z] = act(A[z]*B[z] + bias[z]) (+ resid[z])
// fp32 in/out, bf16 WMMA math.  Block: 256 thr = 8 waves -> 64x64 tile.
// Wave tile 16x32 -> 2 x v_wmma_f32_16x16x32_bf16 per K step (shared A frag).
// Software-pipelined: register-staged next tile + double-buffered LDS,
// one barrier per K step; global-load latency hidden behind WMMA.
// =====================================================================
__global__ __launch_bounds__(256)
void gemm_wmma_bf16(const float* __restrict__ A, long long sA, int lda,
                    const float* __restrict__ B, long long sB, int ldb,
                    float* __restrict__ C, long long sC, int ldc,
                    const float* __restrict__ bias, long long sBias,
                    const float* __restrict__ resid,
                    int M, int N, int K, int act) {
  const int tid  = threadIdx.x;
  const int lane = tid & 31;
  const int wave = tid >> 5;
  const int wm   = wave >> 1;     // 0..3 : wave row (16 rows)
  const int wn   = wave & 1;      // 0..1 : wave col (32 cols)
  const int m0   = blockIdx.y * 64;
  const int n0   = blockIdx.x * 64;
  const int z    = blockIdx.z;

  A += (long long)z * sA;
  B += (long long)z * sB;
  C += (long long)z * sC;
  if (bias)  bias  += (long long)z * sBias;
  if (resid) resid += (long long)z * sC;

  // pitch 40 bf16 = 80B : 16B-aligned rows, staggered banks; double buffered
  __shared__ __attribute__((aligned(16))) __bf16 As[2][64][40];  // [m][k]
  __shared__ __attribute__((aligned(16))) __bf16 Bt[2][64][40];  // [n][k]

  floatx8 acc0 = {}, acc1 = {};

  const int arow = tid >> 2;        // 0..63 (M row)
  const int acol = (tid & 3) * 8;   // 0,8,16,24 (K col)
  const int bn   = tid & 63;        // 0..63 (N col)
  const int bk   = (tid >> 6) * 8;  // 0,8,16,24 (K row)

  const int mr = lane & 15;         // index within 16
  const int hi = lane >> 4;         // half-select per ISA frag layout

  const bool fullM = (m0 + 64 <= M);
  const bool fullN = (n0 + 64 <= N);
  const int kSteps = (K + 31) >> 5;

  float ra[8], rb[8];               // register-staged next tile (fp32)

  auto loadTiles = [&](int k0) {
    const int gr = m0 + arow;
    if (fullM && (k0 + 32 <= K)) {
      const float4 v0 = *(const float4*)&A[(long long)gr * lda + k0 + acol];
      const float4 v1 = *(const float4*)&A[(long long)gr * lda + k0 + acol + 4];
      ra[0] = v0.x; ra[1] = v0.y; ra[2] = v0.z; ra[3] = v0.w;
      ra[4] = v1.x; ra[5] = v1.y; ra[6] = v1.z; ra[7] = v1.w;
    } else {
#pragma unroll
      for (int j = 0; j < 8; ++j) {
        const int gk = k0 + acol + j;
        ra[j] = (gr < M && gk < K) ? A[(long long)gr * lda + gk] : 0.0f;
      }
    }
    const int gc = n0 + bn;
    if (fullN && (k0 + 32 <= K)) {
#pragma unroll
      for (int j = 0; j < 8; ++j)   // coalesced: 64 consecutive cols per row
        rb[j] = B[(long long)(k0 + bk + j) * ldb + gc];
    } else {
#pragma unroll
      for (int j = 0; j < 8; ++j) {
        const int gk = k0 + bk + j;
        rb[j] = (gk < K && gc < N) ? B[(long long)gk * ldb + gc] : 0.0f;
      }
    }
  };
  auto storeTiles = [&](int buf) {
    bf16x8 pa, pb;
#pragma unroll
    for (int j = 0; j < 8; ++j) { pa[j] = (__bf16)ra[j]; pb[j] = (__bf16)rb[j]; }
    *(bf16x8*)&As[buf][arow][acol] = pa;
    *(bf16x8*)&Bt[buf][bn][bk] = pb;
  };

  loadTiles(0);
  storeTiles(0);
  __syncthreads();

  for (int ks = 0; ks < kSteps; ++ks) {
    const int cur = ks & 1;
    // issue next tile's global loads; waits land after the WMMAs
    if (ks + 1 < kSteps) loadTiles((ks + 1) << 5);

    // ---- fragments via contiguous b128 LDS loads ----
    const int arw = wm * 16 + mr;
    const bf16x8 a0 = *(const bf16x8*)&As[cur][arw][hi * 8];
    const bf16x8 a1 = *(const bf16x8*)&As[cur][arw][16 + hi * 8];
    const bf16x16 af = __builtin_shufflevector(
        a0, a1, 0, 1, 2, 3, 4, 5, 6, 7, 8, 9, 10, 11, 12, 13, 14, 15);

    const int c0 = wn * 32 + mr;
    const bf16x8 b00 = *(const bf16x8*)&Bt[cur][c0][hi * 8];
    const bf16x8 b01 = *(const bf16x8*)&Bt[cur][c0][16 + hi * 8];
    const bf16x16 bf0 = __builtin_shufflevector(
        b00, b01, 0, 1, 2, 3, 4, 5, 6, 7, 8, 9, 10, 11, 12, 13, 14, 15);
    const int c1 = c0 + 16;
    const bf16x8 b10 = *(const bf16x8*)&Bt[cur][c1][hi * 8];
    const bf16x8 b11 = *(const bf16x8*)&Bt[cur][c1][16 + hi * 8];
    const bf16x16 bf1 = __builtin_shufflevector(
        b10, b11, 0, 1, 2, 3, 4, 5, 6, 7, 8, 9, 10, 11, 12, 13, 14, 15);

    acc0 = __builtin_amdgcn_wmma_f32_16x16x32_bf16(
        false, af, false, bf0, (short)0, acc0, false, false);
    acc1 = __builtin_amdgcn_wmma_f32_16x16x32_bf16(
        false, af, false, bf1, (short)0, acc1, false, false);

    if (ks + 1 < kSteps) storeTiles(1 - cur);   // fill other buffer
    __syncthreads();
  }

  // ---- epilogue: C/D layout -> VGPR r: M = hi*8 + r, N = mr ----
  const int colA = n0 + wn * 32 + mr;
  const int colB = colA + 16;
  const float bvA = (bias && colA < N) ? bias[colA] : 0.0f;
  const float bvB = (bias && colB < N) ? bias[colB] : 0.0f;
#pragma unroll
  for (int r = 0; r < 8; ++r) {
    const int row = m0 + wm * 16 + hi * 8 + r;
    if (row < M) {
      if (colA < N) {
        float v = acc0[r] + bvA;
        if (act == 1) v = 0.5f * v * (1.0f + erff(v * 0.70710678118f));
        if (resid) v += resid[(long long)row * ldc + colA];
        C[(long long)row * ldc + colA] = v;
      }
      if (colB < N) {
        float v = acc1[r] + bvB;
        if (act == 1) v = 0.5f * v * (1.0f + erff(v * 0.70710678118f));
        if (resid) v += resid[(long long)row * ldc + colB];
        C[(long long)row * ldc + colB] = v;
      }
    }
  }
}

// ---------------- embedding: tokens + positions ----------------
__global__ __launch_bounds__(256)
void embed_kernel(const int* __restrict__ ids, const float* __restrict__ tok,
                  const float* __restrict__ pos, const int* __restrict__ hist,
                  float* __restrict__ h) {
  const int s = blockIdx.x;
  const int id = ids[s];
  const int hoff = hist[0];
  for (int c = threadIdx.x; c < D_DIM; c += 256)
    h[(long long)s * D_DIM + c] =
        tok[(long long)id * D_DIM + c] + pos[(long long)(hoff + s) * D_DIM + c];
}

// ---------------- layernorm over D=768 ----------------
__global__ __launch_bounds__(256)
void layernorm(const float* __restrict__ x, const float* __restrict__ w,
               const float* __restrict__ b, float* __restrict__ y) {
  const int r = blockIdx.x;
  const float* xr = x + (long long)r * D_DIM;
  float* yr = y + (long long)r * D_DIM;
  __shared__ float r1[256], r2[256];
  const int tid = threadIdx.x;
  float s = 0.f, s2 = 0.f;
  for (int i = tid; i < D_DIM; i += 256) { float v = xr[i]; s += v; s2 += v * v; }
  r1[tid] = s; r2[tid] = s2; __syncthreads();
  for (int st = 128; st > 0; st >>= 1) {
    if (tid < st) { r1[tid] += r1[tid + st]; r2[tid] += r2[tid + st]; }
    __syncthreads();
  }
  const float mean = r1[0] * (1.0f / D_DIM);
  const float var  = r2[0] * (1.0f / D_DIM) - mean * mean;
  const float inv  = rsqrtf(var + 1e-5f);
  for (int i = tid; i < D_DIM; i += 256)
    yr[i] = (xr[i] - mean) * inv * w[i] + b[i];
}

// ---------------- softmax rows, optional causal mask ----------------
__global__ __launch_bounds__(256)
void softmax_mask(float* __restrict__ sc, int NK, int causal,
                  const int* __restrict__ mflag, const int* __restrict__ hist) {
  const int row = blockIdx.x;                      // query index 0..447
  float* p = sc + ((long long)blockIdx.y * S_LEN + row) * NK;
  const float mf = causal ? (float)mflag[0] : 0.0f;
  const int hoff = causal ? hist[0] : 0;
  __shared__ float red[256];
  const int tid = threadIdx.x;

  float mx = -1e30f;
  for (int t = tid; t < NK; t += 256) {
    float v = p[t];
    if (causal && t > row + hoff) v -= 128.0f * mf;
    mx = fmaxf(mx, v);
  }
  red[tid] = mx; __syncthreads();
  for (int s = 128; s > 0; s >>= 1) {
    if (tid < s) red[tid] = fmaxf(red[tid], red[tid + s]);
    __syncthreads();
  }
  mx = red[0]; __syncthreads();

  float sum = 0.f;
  for (int t = tid; t < NK; t += 256) {
    float v = p[t];
    if (causal && t > row + hoff) v -= 128.0f * mf;
    v = __expf(v - mx);
    p[t] = v; sum += v;
  }
  red[tid] = sum; __syncthreads();
  for (int s = 128; s > 0; s >>= 1) {
    if (tid < s) red[tid] += red[tid + s];
    __syncthreads();
  }
  const float inv = 1.0f / red[0];
  for (int t = tid; t < NK; t += 256) p[t] *= inv;
}

// ---------------- K-cache transpose: new_k[h][e][t] = kbuf[t][h*64+e] -----
// 32x32 tile through LDS; load side uses gfx1250 async global->LDS DMA
// (ASYNCcnt) when available; both global read and write are coalesced.
__global__ __launch_bounds__(256)
void transpose_k(const float* __restrict__ src, float* __restrict__ dst) {
  const int h  = blockIdx.z;
  const int t0 = blockIdx.x * 32;   // 448/32 = 14
  const int e0 = blockIdx.y * 32;   // 64/32 = 2
  __shared__ __attribute__((aligned(16))) float tile[32][36];
  const int tid = threadIdx.x;
  const int lr = tid >> 3;          // 0..31  (t within tile)
  const int lc = (tid & 7) * 4;     // 0..28  (e within tile, 16B chunks)
  const float* gsrc = src + (long long)(t0 + lr) * D_DIM + h * HD_DIM + e0 + lc;
#if HAVE_ASYNC_LDS
  __builtin_amdgcn_global_load_async_to_lds_b128(
      (__attribute__((address_space(1))) v4i_t*)gsrc,
      (__attribute__((address_space(3))) v4i_t*)&tile[lr][lc], 0, 0);
  __builtin_amdgcn_s_wait_asynccnt(0);
#else
  *(float4*)&tile[lr][lc] = *(const float4*)gsrc;
#endif
  __syncthreads();
  const int oe = tid >> 3;          // 0..31  (e row of output)
  const int ot = (tid & 7) * 4;     // 0..28  (t, coalesced across threads)
  float4 v;
  v.x = tile[ot + 0][oe]; v.y = tile[ot + 1][oe];
  v.z = tile[ot + 2][oe]; v.w = tile[ot + 3][oe];
  *(float4*)&dst[((long long)h * HD_DIM + e0 + oe) * S_LEN + t0 + ot] = v;
}

// new_v[h][t][e] = vbuf[t][h*64+e]   (so new_v is directly B of A*V)
__global__ __launch_bounds__(256)
void write_v_cache(const float* __restrict__ vbuf, float* __restrict__ dst) {
  const int idx = blockIdx.x * 256 + threadIdx.x;
  if (idx >= S_LEN * D_DIM) return;
  const int t = idx / D_DIM, c = idx % D_DIM;
  const int h = c >> 6, e = c & 63;
  dst[((long long)h * S_LEN + t) * HD_DIM + e] = vbuf[idx];
}

// ---------------- logits GEMV: [1,768] x [768,51865] (bandwidth-bound) ----
__global__ __launch_bounds__(256)
void gemv_logits(const float* __restrict__ f, const float* __restrict__ W,
                 float* __restrict__ out) {
  __shared__ float fs[D_DIM];
  for (int i = threadIdx.x; i < D_DIM; i += 256) fs[i] = f[i];
  __syncthreads();
  const int n = blockIdx.x * 256 + threadIdx.x;
  if (n < V_SIZE) {
    float acc = 0.f;
#pragma unroll 4
    for (int d = 0; d < D_DIM; ++d) acc += fs[d] * W[(long long)d * V_SIZE + n];
    out[n] = acc;
  }
}

__global__ __launch_bounds__(256)
void argmax_stage1(const float* __restrict__ logits, float* __restrict__ pmax,
                   int* __restrict__ pidx, int n) {
  const int tid = threadIdx.x;
  float best = -1e30f; int bi = 0;
  for (int i = blockIdx.x * 256 + tid; i < n; i += gridDim.x * 256) {
    float v = logits[i];
    if (v > best) { best = v; bi = i; }
  }
  __shared__ float rv[256]; __shared__ int ri[256];
  rv[tid] = best; ri[tid] = bi; __syncthreads();
  for (int s = 128; s > 0; s >>= 1) {
    if (tid < s && rv[tid + s] > rv[tid]) { rv[tid] = rv[tid + s]; ri[tid] = ri[tid + s]; }
    __syncthreads();
  }
  if (tid == 0) { pmax[blockIdx.x] = rv[0]; pidx[blockIdx.x] = ri[0]; }
}

__global__ __launch_bounds__(256)
void argmax_stage2(const float* __restrict__ pmax, const int* __restrict__ pidx,
                   int nb, int* __restrict__ outIdx, int* __restrict__ outKv,
                   const int* __restrict__ hist) {
  const int tid = threadIdx.x;
  __shared__ float rv[256]; __shared__ int ri[256];
  float best = -1e30f; int bi = 0;
  for (int i = tid; i < nb; i += 256)
    if (pmax[i] > best) { best = pmax[i]; bi = pidx[i]; }
  rv[tid] = best; ri[tid] = bi; __syncthreads();
  for (int s = 128; s > 0; s >>= 1) {
    if (tid < s && rv[tid + s] > rv[tid]) { rv[tid] = rv[tid + s]; ri[tid] = ri[tid + s]; }
    __syncthreads();
  }
  if (tid == 0) { outIdx[0] = ri[0]; outKv[0] = hist[0] + S_LEN; }
}

// =====================================================================
extern "C" void kernel_launch(void* const* d_in, const int* in_sizes, int n_in,
                              void* d_out, int out_size, void* d_ws, size_t ws_size,
                              hipStream_t stream) {
  (void)in_sizes; (void)n_in; (void)out_size; (void)ws_size;
  const int*   input_ids = (const int*)  d_in[0];
  const float* tok_emb   = (const float*)d_in[1];
  const float* pos_emb   = (const float*)d_in[2];
  const float* cross_k   = (const float*)d_in[5];
  const float* cross_v   = (const float*)d_in[6];
  const float* sa_ln_w = (const float*)d_in[7],  *sa_ln_b = (const float*)d_in[8];
  const float* sa_q_w  = (const float*)d_in[9],  *sa_q_b  = (const float*)d_in[10];
  const float* sa_k_w  = (const float*)d_in[11];
  const float* sa_v_w  = (const float*)d_in[12], *sa_v_b  = (const float*)d_in[13];
  const float* sa_o_w  = (const float*)d_in[14], *sa_o_b  = (const float*)d_in[15];
  const float* ca_ln_w = (const float*)d_in[16], *ca_ln_b = (const float*)d_in[17];
  const float* ca_q_w  = (const float*)d_in[18], *ca_q_b  = (const float*)d_in[19];
  const float* ca_o_w  = (const float*)d_in[20], *ca_o_b  = (const float*)d_in[21];
  const float* fin_ln_w= (const float*)d_in[22], *fin_ln_b= (const float*)d_in[23];
  const float* fc1_w   = (const float*)d_in[24], *fc1_b   = (const float*)d_in[25];
  const float* fc2_w   = (const float*)d_in[26], *fc2_b   = (const float*)d_in[27];
  const float* out_ln_w= (const float*)d_in[28], *out_ln_b= (const float*)d_in[29];
  const float* proj_w  = (const float*)d_in[30];
  const int* history_len = (const int*)d_in[31];
  const int* mask_flag   = (const int*)d_in[33];

  float* out_f = (float*)d_out;
  float* out_newK = out_f;                                            // [L,H,64,448]
  float* out_newV = out_f + (long long)L_NUM * H_NUM * HD_DIM * S_LEN;// [L,H,448,64]
  float* out_logits = out_newV + (long long)L_NUM * H_NUM * HD_DIM * S_LEN;
  int*   out_ids = (int*)(out_logits + V_SIZE);
  int*   out_kv  = out_ids + 1;

  // workspace carve-up (floats)
  float* ws = (float*)d_ws;
  const long long SD = (long long)S_LEN * D_DIM;   // 344064
  float* h0    = ws;            ws += SD;
  float* h1    = ws;            ws += SD;
  float* lnbuf = ws;            ws += SD;
  float* qbuf  = ws;            ws += SD;
  float* kbuf  = ws;            ws += SD;
  float* vbuf  = ws;            ws += SD;
  float* abuf  = ws;            ws += SD;
  float* ffbuf = ws;            ws += (long long)S_LEN * FF_DIM;
  float* sbuf  = ws;            ws += (long long)H_NUM * S_LEN * TENC_LEN;
  float* fvec  = ws;            ws += D_DIM;
  float* pmax  = ws;            ws += 128;
  int*   pidx  = (int*)ws;

  const dim3 blk(256);
  auto gemm = [&](const float* A, long long sA, int lda,
                  const float* B, long long sB, int ldb,
                  float* C, long long sC, int ldc,
                  const float* bias, long long sBias, const float* resid,
                  int M, int N, int K, int batch, int act) {
    dim3 grid((N + 63) / 64, (M + 63) / 64, batch);
    gemm_wmma_bf16<<<grid, blk, 0, stream>>>(A, sA, lda, B, sB, ldb, C, sC, ldc,
                                             bias, sBias, resid, M, N, K, act);
  };

  // --- embeddings ---
  embed_kernel<<<dim3(S_LEN), blk, 0, stream>>>(input_ids, tok_emb, pos_emb,
                                                history_len, h0);
  float* hc = h0;  // current residual stream
  float* ha = h1;

  const long long W_HDHD = (long long)H_NUM * D_DIM * HD_DIM;   // 589824 per layer
  const long long CK = (long long)H_NUM * HD_DIM * TENC_LEN;    // 1152000 per layer
  const int cpBlocks = (int)((SD + 255) / 256);

  for (int l = 0; l < L_NUM; ++l) {
    // ===== self attention =====
    layernorm<<<dim3(S_LEN), blk, 0, stream>>>(hc, sa_ln_w + l * D_DIM,
                                               sa_ln_b + l * D_DIM, lnbuf);
    // per-head Q/K/V projections: A[448x768] x W_h[768x64]
    gemm(lnbuf, 0, D_DIM, sa_q_w + l * W_HDHD, (long long)D_DIM * HD_DIM, HD_DIM,
         qbuf, HD_DIM, D_DIM, sa_q_b + l * D_DIM, HD_DIM, nullptr,
         S_LEN, HD_DIM, D_DIM, H_NUM, 0);
    gemm(lnbuf, 0, D_DIM, sa_k_w + l * W_HDHD, (long long)D_DIM * HD_DIM, HD_DIM,
         kbuf, HD_DIM, D_DIM, nullptr, 0, nullptr,
         S_LEN, HD_DIM, D_DIM, H_NUM, 0);
    gemm(lnbuf, 0, D_DIM, sa_v_w + l * W_HDHD, (long long)D_DIM * HD_DIM, HD_DIM,
         vbuf, HD_DIM, D_DIM, sa_v_b + l * D_DIM, HD_DIM, nullptr,
         S_LEN, HD_DIM, D_DIM, H_NUM, 0);

    float* newK_l = out_newK + (long long)l * H_NUM * HD_DIM * S_LEN; // [H,64,448]
    float* newV_l = out_newV + (long long)l * H_NUM * S_LEN * HD_DIM; // [H,448,64]
    transpose_k<<<dim3(S_LEN / 32, HD_DIM / 32, H_NUM), blk, 0, stream>>>(kbuf,
                                                                          newK_l);
    write_v_cache<<<dim3(cpBlocks), blk, 0, stream>>>(vbuf, newV_l);

    // scores[h] = Q_h[448x64] x K_h[64x448]  (newK used as B directly)
    gemm(qbuf, HD_DIM, D_DIM, newK_l, (long long)HD_DIM * S_LEN, S_LEN,
         sbuf, (long long)S_LEN * S_LEN, S_LEN, nullptr, 0, nullptr,
         S_LEN, S_LEN, HD_DIM, H_NUM, 0);
    softmax_mask<<<dim3(S_LEN, H_NUM), blk, 0, stream>>>(sbuf, S_LEN, 1,
                                                         mask_flag, history_len);
    // attn[h] = P_h[448x448] x V_h[448x64]
    gemm(sbuf, (long long)S_LEN * S_LEN, S_LEN, newV_l, (long long)S_LEN * HD_DIM,
         HD_DIM, abuf, HD_DIM, D_DIM, nullptr, 0, nullptr,
         S_LEN, HD_DIM, S_LEN, H_NUM, 0);
    // O-projection + residual:  [448x768] x [768x768]
    gemm(abuf, 0, D_DIM, sa_o_w + l * W_HDHD, 0, D_DIM,
         ha, 0, D_DIM, sa_o_b + l * D_DIM, 0, hc,
         S_LEN, D_DIM, D_DIM, 1, 0);
    { float* t = hc; hc = ha; ha = t; }

    // ===== cross attention =====
    layernorm<<<dim3(S_LEN), blk, 0, stream>>>(hc, ca_ln_w + l * D_DIM,
                                               ca_ln_b + l * D_DIM, lnbuf);
    gemm(lnbuf, 0, D_DIM, ca_q_w + l * W_HDHD, (long long)D_DIM * HD_DIM, HD_DIM,
         qbuf, HD_DIM, D_DIM, ca_q_b + l * D_DIM, HD_DIM, nullptr,
         S_LEN, HD_DIM, D_DIM, H_NUM, 0);
    // scores[h] = Q_h[448x64] x crossK_h[64x1500]
    gemm(qbuf, HD_DIM, D_DIM, cross_k + l * CK, (long long)HD_DIM * TENC_LEN,
         TENC_LEN, sbuf, (long long)S_LEN * TENC_LEN, TENC_LEN, nullptr, 0, nullptr,
         S_LEN, TENC_LEN, HD_DIM, H_NUM, 0);
    softmax_mask<<<dim3(S_LEN, H_NUM), blk, 0, stream>>>(sbuf, TENC_LEN, 0,
                                                         mask_flag, history_len);
    // attn[h] = P_h[448x1500] x crossV_h[1500x64]
    gemm(sbuf, (long long)S_LEN * TENC_LEN, TENC_LEN, cross_v + l * CK,
         (long long)TENC_LEN * HD_DIM, HD_DIM, abuf, HD_DIM, D_DIM,
         nullptr, 0, nullptr, S_LEN, HD_DIM, TENC_LEN, H_NUM, 0);
    gemm(abuf, 0, D_DIM, ca_o_w + l * W_HDHD, 0, D_DIM,
         ha, 0, D_DIM, ca_o_b + l * D_DIM, 0, hc,
         S_LEN, D_DIM, D_DIM, 1, 0);
    { float* t = hc; hc = ha; ha = t; }

    // ===== FFN =====
    layernorm<<<dim3(S_LEN), blk, 0, stream>>>(hc, fin_ln_w + l * D_DIM,
                                               fin_ln_b + l * D_DIM, lnbuf);
    gemm(lnbuf, 0, D_DIM, fc1_w + (long long)l * D_DIM * FF_DIM, 0, FF_DIM,
         ffbuf, 0, FF_DIM, fc1_b + l * FF_DIM, 0, nullptr,
         S_LEN, FF_DIM, D_DIM, 1, /*gelu*/1);
    gemm(ffbuf, 0, FF_DIM, fc2_w + (long long)l * FF_DIM * D_DIM, 0, D_DIM,
         ha, 0, D_DIM, fc2_b + l * D_DIM, 0, hc,
         S_LEN, D_DIM, FF_DIM, 1, 0);
    { float* t = hc; hc = ha; ha = t; }
  }

  // ===== final LN on last token + logits + argmax =====
  layernorm<<<dim3(1), blk, 0, stream>>>(hc + (long long)(S_LEN - 1) * D_DIM,
                                         out_ln_w, out_ln_b, fvec);
  gemv_logits<<<dim3((V_SIZE + 255) / 256), blk, 0, stream>>>(fvec, proj_w,
                                                              out_logits);
  argmax_stage1<<<dim3(64), blk, 0, stream>>>(out_logits, pmax, pidx, V_SIZE);
  argmax_stage2<<<dim3(1), blk, 0, stream>>>(pmax, pidx, 64, out_ids, out_kv,
                                             history_len);
}
